// peepLSTM_69028714381452
// MI455X (gfx1250) — matrix operand order repaired and checked by reading, not compiled, not run on hardware
//
#include <hip/hip_runtime.h>
#include <hip/hip_bf16.h>

#define B_ 256
#define S_ 512
#define I_ 256
#define H_ 1024
#define C_ 128

typedef __attribute__((ext_vector_type(16))) __bf16 v16bf;
typedef __attribute__((ext_vector_type(8)))  __bf16 v8bf;
typedef __attribute__((ext_vector_type(8)))  float  v8f;

// ---------- helpers ----------

static __device__ __forceinline__ unsigned short f2bf(float f) {
  unsigned int u = __float_as_uint(f);
  unsigned int r = u + 0x7fffu + ((u >> 16) & 1u);
  return (unsigned short)(r >> 16);
}

static __device__ __forceinline__ float sigmoidf(float x) {
  return 1.0f / (1.0f + __expf(-x));
}

// A-matrix (16x32 bf16, M x K) per-lane load:
// lane = hi*16 + m ; elems 0..15 -> K = k0 + hi*16 .. +15 (matches ldB)
static __device__ __forceinline__ v16bf ldA(const unsigned short* rowbase, int k0, int hi) {
  return *(const v16bf*)(rowbase + k0 + hi * 16);
}

// B-matrix (32x16 bf16, K x N): lane hi*16+n holds K = k0 + hi*16 .. +15 of column n,
// i.e. one contiguous 32B chunk of row-major W[h0+n][*]. Works for global or LDS pointers.
static __device__ __forceinline__ v16bf ldB(const unsigned short* rowbase, int k0, int hi) {
  return *(const v16bf*)(rowbase + k0 + hi * 16);
}

static __device__ __forceinline__ v8f wmma_bf16(v16bf a, v16bf b, v8f c) {
  return __builtin_amdgcn_wmma_f32_16x16x32_bf16(false, a, false, b, (short)0, c, false, false);
}

// CDNA5 async direct global->LDS copy (16B), tracked by ASYNCcnt (ISA 08_async_tensor §4).
// LDS byte address = low 32 bits of the flat shared-pointer address.
static __device__ __forceinline__ void async_cp16(unsigned short* lds_dst, const unsigned short* gsrc) {
  unsigned ldsaddr = (unsigned)(unsigned long long)lds_dst;
  asm volatile("global_load_async_to_lds_b128 %0, %1, off"
               :: "v"(ldsaddr), "v"(gsrc)
               : "memory");
}

static __device__ __forceinline__ void async_wait0() {
  asm volatile("s_wait_asynccnt 0" ::: "memory");
}

// ---------- kernels ----------

// gather embedding + convert to bf16, layout [s][b][i]
__global__ __launch_bounds__(256) void embed_kernel(const int* __restrict__ x,
                                                    const float* __restrict__ emb,
                                                    unsigned short* __restrict__ xt) {
  size_t idx = (size_t)blockIdx.x * blockDim.x + threadIdx.x;  // < S*B*I (exact grid)
  int i = (int)(idx % I_);
  size_t sb = idx / I_;
  int b = (int)(sb % B_);
  int s = (int)(sb / B_);
  int tok = x[(size_t)b * S_ + s];
  xt[idx] = f2bf(emb[(size_t)tok * I_ + i]);
}

__global__ __launch_bounds__(256) void cvt_kernel(const float* __restrict__ in,
                                                  unsigned short* __restrict__ out, int n) {
  int i = blockIdx.x * blockDim.x + threadIdx.x;
  if (i < n) out[i] = f2bf(in[i]);
}

__global__ __launch_bounds__(256) void zero_c_kernel(float* __restrict__ c,
                                                     unsigned short* __restrict__ cb) {
  int i = blockIdx.x * blockDim.x + threadIdx.x;  // exact grid = B*H
  c[i] = 0.0f;
  cb[i] = 0;
}

// One LSTM time step. Block = 128 threads = 4 waves, covering 1 hidden-tile x 4 batch-tiles.
// Recurrent-gate weights for the block's h-tile are staged into LDS once (async copy) and
// shared by the 4 waves; B-operands then come from ds_load. Gates f,i: K = H (over carry c)
// + I (fused x-projection). Gate c: K = I only. Gate o only when LAST (h never feeds back).
template <bool LAST>
__global__ __launch_bounds__(128) void lstm_step_kernel(
    const unsigned short* __restrict__ c_bf, const float* __restrict__ c_f32,
    const unsigned short* __restrict__ xt_t,               // [B][I] bf16 at time t
    const unsigned short* __restrict__ Wfh_b, const unsigned short* __restrict__ Wih_b,
    const unsigned short* __restrict__ Woh_b,
    const unsigned short* __restrict__ Wfx_b, const unsigned short* __restrict__ Wix_b,
    const unsigned short* __restrict__ Wcx_b, const unsigned short* __restrict__ Wox_b,
    const float* __restrict__ bfv, const float* __restrict__ biv,
    const float* __restrict__ bcv, const float* __restrict__ bov,
    float* __restrict__ cn_f32, unsigned short* __restrict__ cn_bf,
    unsigned short* __restrict__ h_bf) {
  constexpr int NW = LAST ? 3 : 2;
  __shared__ unsigned short sW[NW * 16 * H_];  // 64 KB (96 KB on last step)
  unsigned short* sWf = sW;
  unsigned short* sWi = sW + 16 * H_;
  unsigned short* sWo = sW + (NW - 1) * 16 * H_;  // aliases sWi when !LAST (unused then)

  const int tid = threadIdx.x;
  const int ht = blockIdx.x >> 2;  // 0..63
  const int h0 = ht * 16;

  // Stage the 16 recurrent-weight rows per gate for this h-tile (rows are contiguous).
  {
    const unsigned short* gf = Wfh_b + (size_t)h0 * H_;
    const unsigned short* gi = Wih_b + (size_t)h0 * H_;
    const unsigned short* go = Woh_b + (size_t)h0 * H_;
    for (int ofs = tid * 8; ofs < 16 * H_; ofs += 128 * 8) {
      async_cp16(sWf + ofs, gf + ofs);
      async_cp16(sWi + ofs, gi + ofs);
      if (LAST) async_cp16(sWo + ofs, go + ofs);
    }
    async_wait0();
    __syncthreads();
  }

  const int wave = tid >> 5;
  const int lane = tid & 31;
  const int m  = lane & 15;
  const int hi = lane >> 4;
  const int b0 = ((int)(blockIdx.x & 3) * 4 + wave) * 16;

  v8f accF = {}, accI = {}, accC = {}, accO = {};

  const unsigned short* arow = c_bf + (size_t)(b0 + m) * H_;  // A rows: carry c (global)
  const unsigned short* swf  = sWf + m * H_;                  // B rows: LDS-resident
  const unsigned short* swi  = sWi + m * H_;
  const unsigned short* swo  = sWo + m * H_;

  // recurrent part: K over H — branch-free inner loop (LAST is compile-time)
  for (int k0 = 0; k0 < H_; k0 += 32) {
    __builtin_prefetch(arow + k0 + 64, 0, 3);
    v16bf a = ldA(arow, k0, hi);
    accF = wmma_bf16(a, ldB(swf, k0, hi), accF);
    accI = wmma_bf16(a, ldB(swi, k0, hi), accI);
    if (LAST) accO = wmma_bf16(a, ldB(swo, k0, hi), accO);
  }

  // fused x-projection part: K over I (small weight matrices, L2-resident -> global reads)
  const unsigned short* xrow = xt_t + (size_t)(b0 + m) * I_;
  const unsigned short* wfx  = Wfx_b + (size_t)(h0 + m) * I_;
  const unsigned short* wix  = Wix_b + (size_t)(h0 + m) * I_;
  const unsigned short* wcx  = Wcx_b + (size_t)(h0 + m) * I_;
  const unsigned short* wox  = Wox_b + (size_t)(h0 + m) * I_;
  for (int k0 = 0; k0 < I_; k0 += 32) {
    v16bf a = ldA(xrow, k0, hi);
    accF = wmma_bf16(a, ldB(wfx, k0, hi), accF);
    accI = wmma_bf16(a, ldB(wix, k0, hi), accI);
    accC = wmma_bf16(a, ldB(wcx, k0, hi), accC);
    if (LAST) accO = wmma_bf16(a, ldB(wox, k0, hi), accO);
  }

  // elementwise update; C/D layout: elem r of lane (hi*16+n) is row b0+r+8*hi, col h0+n
  const int h = h0 + m;
  const float Bf = bfv[h], Bi = biv[h], Bc = bcv[h];
  const float Bo = LAST ? bov[h] : 0.0f;
#pragma unroll
  for (int r = 0; r < 8; ++r) {
    const int b = b0 + r + hi * 8;
    const size_t idx = (size_t)b * H_ + h;
    float f = sigmoidf(accF[r] + Bf);
    float i = sigmoidf(accI[r] + Bi);
    float g = sigmoidf(accC[r] + Bc);
    float cold = c_f32[idx];
    float cn = g * i + cold * f;
    cn_f32[idx] = cn;          // fp32 carry for precision
    cn_bf[idx] = f2bf(cn);     // bf16 shadow: next step's WMMA A operand
    if (LAST) {
      float o = sigmoidf(accO[r] + Bo);
      h_bf[idx] = f2bf(tanhf(cn) * o);
    }
  }
}

// p = h_T @ Wph^T + bp  (256 x 128, K = 1024), one wave per 16x16 tile
__global__ __launch_bounds__(128) void proj_kernel(const unsigned short* __restrict__ h_bf,
                                                   const unsigned short* __restrict__ Wph_b,
                                                   const float* __restrict__ bp,
                                                   float* __restrict__ pout) {
  const int wid  = (blockIdx.x * blockDim.x + threadIdx.x) >> 5;  // 0..127
  const int lane = threadIdx.x & 31;
  const int m = lane & 15, hi = lane >> 4;
  const int b0 = (wid >> 3) * 16;  // 16 row tiles
  const int n0 = (wid & 7) * 16;   // 8 class tiles
  v8f acc = {};
  const unsigned short* arow = h_bf + (size_t)(b0 + m) * H_;
  const unsigned short* brow = Wph_b + (size_t)(n0 + m) * H_;
  for (int k0 = 0; k0 < H_; k0 += 32)
    acc = wmma_bf16(ldA(arow, k0, hi), ldB(brow, k0, hi), acc);
  const int cls = n0 + m;
  const float bias = bp[cls];
#pragma unroll
  for (int r = 0; r < 8; ++r)
    pout[(size_t)(b0 + r + hi * 8) * C_ + cls] = acc[r] + bias;
}

// log-softmax over 128 classes, one wave32 per row (4 classes per lane)
__global__ __launch_bounds__(128) void lsm_kernel(const float* __restrict__ p,
                                                  float* __restrict__ out) {
  const int row  = (blockIdx.x * blockDim.x + threadIdx.x) >> 5;  // 0..255
  const int lane = threadIdx.x & 31;
  const float* r = p + (size_t)row * C_;
  float v[4];
  float mx = -1e30f;
#pragma unroll
  for (int j = 0; j < 4; ++j) { v[j] = r[lane + 32 * j]; mx = fmaxf(mx, v[j]); }
  for (int o = 16; o > 0; o >>= 1) mx = fmaxf(mx, __shfl_xor(mx, o, 32));
  float s = 0.0f;
#pragma unroll
  for (int j = 0; j < 4; ++j) s += __expf(v[j] - mx);
  for (int o = 16; o > 0; o >>= 1) s += __shfl_xor(s, o, 32);
  const float lse = mx + logf(s);
#pragma unroll
  for (int j = 0; j < 4; ++j) out[(size_t)row * C_ + lane + 32 * j] = v[j] - lse;
}

// ---------- host ----------

extern "C" void kernel_launch(void* const* d_in, const int* in_sizes, int n_in,
                              void* d_out, int out_size, void* d_ws, size_t ws_size,
                              hipStream_t stream) {
  (void)in_sizes; (void)n_in; (void)out_size; (void)ws_size;
  const int*   x    = (const int*)  d_in[0];
  const float* emb  = (const float*)d_in[1];
  const float* Wcx  = (const float*)d_in[2];
  const float* bc   = (const float*)d_in[3];
  const float* Wix  = (const float*)d_in[4];
  const float* Wih  = (const float*)d_in[5];
  const float* bi   = (const float*)d_in[6];
  const float* Wfx  = (const float*)d_in[7];
  const float* Wfh  = (const float*)d_in[8];
  const float* bfp  = (const float*)d_in[9];
  const float* Wox  = (const float*)d_in[10];
  const float* Woh  = (const float*)d_in[11];
  const float* bo   = (const float*)d_in[12];
  const float* Wph  = (const float*)d_in[13];
  const float* bp   = (const float*)d_in[14];

  char* wp = (char*)d_ws;
  auto take = [&](size_t bytes) -> char* {
    char* r = wp;
    wp += (bytes + 255) & ~(size_t)255;
    return r;
  };
  unsigned short* xt_bf = (unsigned short*)take((size_t)S_ * B_ * I_ * 2);  // 64 MB
  unsigned short* Wfh_b = (unsigned short*)take((size_t)H_ * H_ * 2);
  unsigned short* Wih_b = (unsigned short*)take((size_t)H_ * H_ * 2);
  unsigned short* Woh_b = (unsigned short*)take((size_t)H_ * H_ * 2);
  unsigned short* Wfx_b = (unsigned short*)take((size_t)H_ * I_ * 2);
  unsigned short* Wix_b = (unsigned short*)take((size_t)H_ * I_ * 2);
  unsigned short* Wcx_b = (unsigned short*)take((size_t)H_ * I_ * 2);
  unsigned short* Wox_b = (unsigned short*)take((size_t)H_ * I_ * 2);
  unsigned short* Wph_b = (unsigned short*)take((size_t)C_ * H_ * 2);
  float*          cA_f  = (float*)         take((size_t)B_ * H_ * 4);
  float*          cB_f  = (float*)         take((size_t)B_ * H_ * 4);
  unsigned short* cA_b  = (unsigned short*)take((size_t)B_ * H_ * 2);
  unsigned short* cB_b  = (unsigned short*)take((size_t)B_ * H_ * 2);
  unsigned short* h_b   = (unsigned short*)take((size_t)B_ * H_ * 2);
  float*          pbuf  = (float*)         take((size_t)B_ * C_ * 4);

  // Stage inputs: embedding gather->bf16, weights->bf16
  embed_kernel<<<(S_ * B_ * I_) / 256, 256, 0, stream>>>(x, emb, xt_bf);
  cvt_kernel<<<(H_ * H_) / 256, 256, 0, stream>>>(Wfh, Wfh_b, H_ * H_);
  cvt_kernel<<<(H_ * H_) / 256, 256, 0, stream>>>(Wih, Wih_b, H_ * H_);
  cvt_kernel<<<(H_ * H_) / 256, 256, 0, stream>>>(Woh, Woh_b, H_ * H_);
  cvt_kernel<<<(H_ * I_) / 256, 256, 0, stream>>>(Wfx, Wfx_b, H_ * I_);
  cvt_kernel<<<(H_ * I_) / 256, 256, 0, stream>>>(Wix, Wix_b, H_ * I_);
  cvt_kernel<<<(H_ * I_) / 256, 256, 0, stream>>>(Wcx, Wcx_b, H_ * I_);
  cvt_kernel<<<(H_ * I_) / 256, 256, 0, stream>>>(Wox, Wox_b, H_ * I_);
  cvt_kernel<<<(C_ * H_) / 256, 256, 0, stream>>>(Wph, Wph_b, C_ * H_);
  zero_c_kernel<<<(B_ * H_) / 256, 256, 0, stream>>>(cA_f, cA_b);

  // Serial recurrence: 512 step kernels, ping-pong c buffers. o-gate only at t = S-1.
  for (int t = 0; t < S_; ++t) {
    const unsigned short* csrc_b = (t & 1) ? cB_b : cA_b;
    const float*          csrc_f = (t & 1) ? cB_f : cA_f;
    unsigned short*       cdst_b = (t & 1) ? cA_b : cB_b;
    float*                cdst_f = (t & 1) ? cA_f : cB_f;
    const unsigned short* xt_t = xt_bf + (size_t)t * B_ * I_;
    if (t == S_ - 1) {
      lstm_step_kernel<true><<<256, 128, 0, stream>>>(
          csrc_b, csrc_f, xt_t, Wfh_b, Wih_b, Woh_b, Wfx_b, Wix_b, Wcx_b, Wox_b,
          bfp, bi, bc, bo, cdst_f, cdst_b, h_b);
    } else {
      lstm_step_kernel<false><<<256, 128, 0, stream>>>(
          csrc_b, csrc_f, xt_t, Wfh_b, Wih_b, Woh_b, Wfx_b, Wix_b, Wcx_b, Wox_b,
          bfp, bi, bc, bo, cdst_f, cdst_b, h_b);
    }
  }

  proj_kernel<<<32, 128, 0, stream>>>(h_b, Wph_b, bp, pbuf);
  lsm_kernel<<<64, 128, 0, stream>>>(pbuf, (float*)d_out);
}
